// PairToSingle_32744830664942
// MI455X (gfx1250) — matrix operand order repaired and checked by reading, not compiled, not run on hardware
//
#include <hip/hip_runtime.h>
#include <cstdint>
#include <cstddef>

// ---------------------------------------------------------------------------
// Types for CDNA5 WMMA (wave32): D(16x16 f32) = A(16x32 f16) x B(32x16 f16) + C
// All operands here are well-scaled, so f16 (10-bit mantissa) beats bf16 at
// identical WMMA throughput.
// ---------------------------------------------------------------------------
typedef _Float16 half_t;
typedef half_t v16h __attribute__((ext_vector_type(16)));
typedef half_t v8h  __attribute__((ext_vector_type(8)));
typedef float  v8f  __attribute__((ext_vector_type(8)));
typedef float  v4f_ __attribute__((ext_vector_type(4)));

#define N_TOK 1024
#define D_SINGLE 384
#define D_PAIR 128
#define NH 16
#define DH 32
#define HD (NH * DH)              // 512
#define JSPLIT 8
#define JRANGE (N_TOK / JSPLIT)   // 128
#define JCHUNK 32
#define NCHUNK (JRANGE / JCHUNK)  // 4

static __device__ inline v8f wmma_f16(v16h a, v16h b, v8f c) {
  return __builtin_amdgcn_wmma_f32_16x16x32_f16(
      /*neg_a=*/false, a, /*neg_b=*/false, b,
      /*c_mod=*/(short)0, c, /*reuse_a=*/false, /*reuse_b=*/false);
}

// A fragment: 16x32 f16 tile, row-major, row stride `stride` (elements).
// lane<16: row=lane, K in {0..7, 16..23}; lane>=16: row=lane-16, K in {8..15, 24..31}.
static __device__ inline v16h load_a_frag(const half_t* __restrict__ base, int stride) {
  const int lane = threadIdx.x & 31;
  const half_t* r = base + (size_t)(lane & 15) * stride + (lane >> 4) * 8;
  v8h lo = *(const v8h*)(r);
  v8h hi = *(const v8h*)(r + 16);
  return __builtin_shufflevector(lo, hi, 0,1,2,3,4,5,6,7,8,9,10,11,12,13,14,15);
}

// B fragment: 32(K) x 16(N), stored N-major as [16 cols][K...], row stride `stride`.
// lane<16: col=lane, K in {0..7, 16..23}; lane>=16: col=lane-16, K in {8..15, 24..31}.
static __device__ inline v16h load_b_frag(const half_t* __restrict__ base, int stride) {
  const int lane = threadIdx.x & 31;
  const half_t* r = base + (size_t)(lane & 15) * stride + (lane >> 4) * 8;
  v8h lo = *(const v8h*)(r);
  v8h hi = *(const v8h*)(r + 16);
  return __builtin_shufflevector(lo, hi, 0,1,2,3,4,5,6,7,8,9,10,11,12,13,14,15);
}

// ---------------------------------------------------------------------------
// Kernel 1: fp32 -> f16 conversions and weight transposes.
//   s_h   [1024][384]           = single_repr
//   wT    [3][512][384]         = Wq^T, Wk^T, Wv^T   (B layout: [out][k])
//   wbT   [16][128]             = Wb^T
//   woT   [384][512]            = Wo^T
// ---------------------------------------------------------------------------
#define SEG_S   (N_TOK * D_SINGLE)        // 393216
#define SEG_W   (D_SINGLE * HD)           // 196608
#define SEG_WB  (NH * D_PAIR)             // 2048
#define SEG_WO  (HD * D_SINGLE)           // 196608
#define PREP_TOTAL (SEG_S + 3 * SEG_W + SEG_WB + SEG_WO)

__global__ __launch_bounds__(256) void prep_kernel(
    const float* __restrict__ single, const float* __restrict__ Wq,
    const float* __restrict__ Wk, const float* __restrict__ Wv,
    const float* __restrict__ Wb, const float* __restrict__ Wo,
    half_t* __restrict__ s_h, half_t* __restrict__ wT,
    half_t* __restrict__ wbT, half_t* __restrict__ woT) {
  int i = blockIdx.x * 256 + threadIdx.x;
  if (i >= PREP_TOTAL) return;
  if (i < SEG_S) { s_h[i] = (half_t)single[i]; return; }
  i -= SEG_S;
  if (i < SEG_W) { int o = i / D_SINGLE, d = i - o * D_SINGLE;
                   wT[i] = (half_t)Wq[d * HD + o]; return; }
  i -= SEG_W;
  if (i < SEG_W) { int o = i / D_SINGLE, d = i - o * D_SINGLE;
                   wT[SEG_W + i] = (half_t)Wk[d * HD + o]; return; }
  i -= SEG_W;
  if (i < SEG_W) { int o = i / D_SINGLE, d = i - o * D_SINGLE;
                   wT[2 * SEG_W + i] = (half_t)Wv[d * HD + o]; return; }
  i -= SEG_W;
  if (i < SEG_WB) { int h = i / D_PAIR, d = i - h * D_PAIR;
                    wbT[i] = (half_t)Wb[d * NH + h]; return; }
  i -= SEG_WB;
  { int c = i / HD, k = i - c * HD;
    woT[i] = (half_t)Wo[k * D_SINGLE + c]; }
}

// ---------------------------------------------------------------------------
// Kernel 2: QKV projection.  One wave per (n-tile, o-tile, matrix).
//   q_g,k_g [h][n][dh] f16 ; vT_g [h][dh][n] f16 (transposed for P.V B-frags)
// ---------------------------------------------------------------------------
__global__ __launch_bounds__(32) void qkv_kernel(
    const half_t* __restrict__ s_h, const half_t* __restrict__ wT,
    const float* __restrict__ bq, const float* __restrict__ bk,
    const float* __restrict__ bv,
    half_t* __restrict__ q_g, half_t* __restrict__ k_g, half_t* __restrict__ vT_g) {
  const int nt = blockIdx.x, ot = blockIdx.y, mat = blockIdx.z;
  const half_t* w = wT + (size_t)mat * SEG_W;
  const float* bias = (mat == 0) ? bq : (mat == 1) ? bk : bv;
  const int n0 = nt * 16, o0 = ot * 16;
  v8f acc = {};
#pragma unroll
  for (int kw = 0; kw < D_SINGLE / 32; ++kw) {
    v16h a = load_a_frag(s_h + (size_t)n0 * D_SINGLE + kw * 32, D_SINGLE);
    v16h b = load_b_frag(w + (size_t)o0 * D_SINGLE + kw * 32, D_SINGLE);
    acc = wmma_f16(a, b, acc);
  }
  const int lane = threadIdx.x & 31;
  const int lhalf = lane >> 4, col = lane & 15;
  const int o = o0 + col, h = o >> 5, dh = o & 31;
  const float bo_ = bias[o];
#pragma unroll
  for (int m = 0; m < 8; ++m) {
    const int n = n0 + m + 8 * lhalf;
    const half_t val = (half_t)(acc[m] + bo_);
    if (mat == 0)      q_g[((size_t)h * N_TOK + n) * DH + dh] = val;
    else if (mat == 1) k_g[((size_t)h * N_TOK + n) * DH + dh] = val;
    else               vT_g[((size_t)h * DH + dh) * N_TOK + n] = val;
  }
}

// ---------------------------------------------------------------------------
// Kernel 3: fused pair-bias + flash attention.
// grid (64 i-tiles, JSPLIT j-splits); 256 threads = 8 waves.
// Per 32-j chunk: bias WMMA (pair fp32->f16 on the fly, NON-TEMPORAL loads so
// the 512 MB single-use stream does not thrash the 192 MB L2 holding q/k/v)
// -> scores WMMA -> online softmax -> P.V WMMA into register accumulators.
// ---------------------------------------------------------------------------
__global__ __launch_bounds__(256) void attn_kernel(
    const float* __restrict__ pair, const half_t* __restrict__ wbT,
    const float* __restrict__ bb,
    const half_t* __restrict__ q_g, const half_t* __restrict__ k_g,
    const half_t* __restrict__ vT_g,
    float* __restrict__ o_part, float* __restrict__ m_part,
    float* __restrict__ l_part) {
  __shared__ float  s_tile[NH][16][JCHUNK];  // scores (bias then +qk)  32 KB
  __shared__ half_t p_tile[NH][16][JCHUNK];  // softmax probs f16       16 KB
  __shared__ float  m_s[NH][16], l_s[NH][16], r_s[NH][16];  //           3 KB

  const int it = blockIdx.x, jb = blockIdx.y;
  const int i0 = it * 16;
  const int tid = threadIdx.x;
  const int wave = tid >> 5, lane = tid & 31;
  const int lrow = lane & 15, lhalf = lane >> 4;
  const float scale = 0.17677669529663687f;  // 1/sqrt(DH)
  const float NEG_INF = -__builtin_inff();

  { const int h = tid >> 4, i = tid & 15; m_s[h][i] = NEG_INF; l_s[h][i] = 0.f; }

  // per-wave O accumulators: tile idx = wave*4+n -> (h = idx>>1, dh-half = idx&1)
  v8f o_frag[4];
#pragma unroll
  for (int n = 0; n < 4; ++n) o_frag[n] = (v8f){};

  for (int c = 0; c < NCHUNK; ++c) {
    const int j0 = jb * JRANGE + c * JCHUNK;

    // ---- phase 1: pair bias. 32 M-tiles of 16 pairs; wave owns 4.
#pragma unroll
    for (int n = 0; n < 4; ++n) {
      const int t = wave * 4 + n;           // pair M-tile
      const int p = t * 16 + lrow;          // pair = ii*32 + jj
      const int ii = p >> 5, jj = p & 31;
      const float* prow = pair + ((size_t)(i0 + ii) * N_TOK + (j0 + jj)) * D_PAIR;
      if (c + 1 < NCHUNK) __builtin_prefetch(prow + (size_t)JCHUNK * D_PAIR, 0, 0);
      v8f acc = {};
#pragma unroll
      for (int kw = 0; kw < D_PAIR / 32; ++kw) {
        const v4f_ f0 = __builtin_nontemporal_load((const v4f_*)(prow + kw * 32 + lhalf * 8));
        const v4f_ f1 = __builtin_nontemporal_load((const v4f_*)(prow + kw * 32 + lhalf * 8 + 4));
        const v4f_ f2 = __builtin_nontemporal_load((const v4f_*)(prow + kw * 32 + 16 + lhalf * 8));
        const v4f_ f3 = __builtin_nontemporal_load((const v4f_*)(prow + kw * 32 + 16 + lhalf * 8 + 4));
        v16h a;
#pragma unroll
        for (int e = 0; e < 4; ++e) {
          a[e]      = (half_t)f0[e];
          a[4 + e]  = (half_t)f1[e];
          a[8 + e]  = (half_t)f2[e];
          a[12 + e] = (half_t)f3[e];
        }
        v16h b = load_b_frag(wbT + kw * 32, D_PAIR);
        acc = wmma_f16(a, b, acc);
      }
      const int h = lane & 15;              // D col = head
      const float bbh = bb[h];
#pragma unroll
      for (int m = 0; m < 8; ++m) {
        const int pp = t * 16 + m + 8 * lhalf;
        s_tile[h][pp >> 5][pp & 31] = (acc[m] + bbh) * scale;
      }
    }
    __syncthreads();

    // ---- phase 2: scores += q.k^T * scale.  32 tiles (16 heads x 2 j-halves).
#pragma unroll
    for (int n = 0; n < 4; ++n) {
      const int idx = wave * 4 + n;
      const int h = idx >> 1, jh = idx & 1;
      v16h a = load_a_frag(q_g + ((size_t)h * N_TOK + i0) * DH, DH);
      v16h b = load_b_frag(k_g + ((size_t)h * N_TOK + j0 + jh * 16) * DH, DH);
      v8f d = wmma_f16(a, b, (v8f){});
      const int col = jh * 16 + (lane & 15);
#pragma unroll
      for (int m = 0; m < 8; ++m)
        s_tile[h][m + 8 * lhalf][col] += d[m] * scale;
    }
    __syncthreads();

    // ---- phase 3: online softmax; one (h,i) row per thread.
    {
      const int h = tid >> 4, i = tid & 15;
      const float mold = m_s[h][i];
      float rmax = NEG_INF;
#pragma unroll
      for (int j = 0; j < JCHUNK; ++j) rmax = fmaxf(rmax, s_tile[h][i][j]);
      const float mnew = fmaxf(mold, rmax);
      const float r = __expf(mold - mnew);
      float lsum = 0.f;
#pragma unroll
      for (int j = 0; j < JCHUNK; ++j) {
        const float pv = __expf(s_tile[h][i][j] - mnew);
        p_tile[h][i][j] = (half_t)pv;
        lsum += pv;
      }
      m_s[h][i] = mnew;
      l_s[h][i] = l_s[h][i] * r + lsum;
      r_s[h][i] = r;
    }
    __syncthreads();

    // ---- phase 4: O = O*r + P.V.  32 tiles (16 heads x 2 dh-halves).
#pragma unroll
    for (int n = 0; n < 4; ++n) {
      const int idx = wave * 4 + n;
      const int h = idx >> 1, dhh = idx & 1;
#pragma unroll
      for (int m = 0; m < 8; ++m) o_frag[n][m] *= r_s[h][m + 8 * lhalf];
      v16h a = load_a_frag(&p_tile[h][0][0], JCHUNK);
      v16h b = load_b_frag(vT_g + ((size_t)h * DH + dhh * 16) * N_TOK + j0, N_TOK);
      o_frag[n] = wmma_f16(a, b, o_frag[n]);
    }
    __syncthreads();
  }

  // ---- write partials (stay RT: re-read immediately by combine, fits L2)
  const int pblk = it * JSPLIT + jb;
#pragma unroll
  for (int n = 0; n < 4; ++n) {
    const int idx = wave * 4 + n;
    const int h = idx >> 1, dhh = idx & 1;
    const int dh = dhh * 16 + (lane & 15);
#pragma unroll
    for (int m = 0; m < 8; ++m) {
      const int M = m + 8 * lhalf;
      o_part[(((size_t)pblk * NH + h) * 16 + M) * DH + dh] = o_frag[n][m];
    }
  }
  {
    const int h = tid >> 4, i = tid & 15;
    m_part[((size_t)pblk * NH + h) * 16 + i] = m_s[h][i];
    l_part[((size_t)pblk * NH + h) * 16 + i] = l_s[h][i];
  }
}

// ---------------------------------------------------------------------------
// Kernel 4: combine JSPLIT partials -> normalized O (f16 [n][h*DH+dh]).
// ---------------------------------------------------------------------------
__global__ __launch_bounds__(256) void combine_kernel(
    const float* __restrict__ o_part, const float* __restrict__ m_part,
    const float* __restrict__ l_part, half_t* __restrict__ o_g) {
  const int idx = blockIdx.x * 256 + threadIdx.x;  // [n][h][dh]
  if (idx >= N_TOK * NH * DH) return;
  const int dh = idx & 31, h = (idx >> 5) & 15, n = idx >> 9;
  const int it = n >> 4, ii = n & 15;
  float mmax = -__builtin_inff();
#pragma unroll
  for (int b = 0; b < JSPLIT; ++b)
    mmax = fmaxf(mmax, m_part[(((size_t)(it * JSPLIT + b)) * NH + h) * 16 + ii]);
  float denom = 0.f, acc = 0.f;
#pragma unroll
  for (int b = 0; b < JSPLIT; ++b) {
    const size_t pb = (size_t)(it * JSPLIT + b);
    const float mb = m_part[(pb * NH + h) * 16 + ii];
    const float w = __expf(mb - mmax);
    denom += l_part[(pb * NH + h) * 16 + ii] * w;
    acc += o_part[((pb * NH + h) * 16 + ii) * DH + dh] * w;
  }
  o_g[(size_t)n * HD + h * DH + dh] = (half_t)(acc / denom);
}

// ---------------------------------------------------------------------------
// Kernel 5: out = single + O @ Wo + bo.  One wave per (n-tile, c-tile), K=512.
// ---------------------------------------------------------------------------
__global__ __launch_bounds__(32) void proj_kernel(
    const half_t* __restrict__ o_g, const half_t* __restrict__ woT,
    const float* __restrict__ single, const float* __restrict__ bo,
    float* __restrict__ out) {
  const int n0 = blockIdx.x * 16, c0 = blockIdx.y * 16;
  v8f acc = {};
#pragma unroll
  for (int kw = 0; kw < HD / 32; ++kw) {
    v16h a = load_a_frag(o_g + (size_t)n0 * HD + kw * 32, HD);
    v16h b = load_b_frag(woT + (size_t)c0 * HD + kw * 32, HD);
    acc = wmma_f16(a, b, acc);
  }
  const int lane = threadIdx.x & 31;
  const int lhalf = lane >> 4, col = lane & 15;
  const int c = c0 + col;
  const float bc = bo[c];
#pragma unroll
  for (int m = 0; m < 8; ++m) {
    const int n = n0 + m + 8 * lhalf;
    out[(size_t)n * D_SINGLE + c] = acc[m] + bc + single[(size_t)n * D_SINGLE + c];
  }
}

// ---------------------------------------------------------------------------
// Host launcher
// ---------------------------------------------------------------------------
extern "C" void kernel_launch(void* const* d_in, const int* in_sizes, int n_in,
                              void* d_out, int out_size, void* d_ws, size_t ws_size,
                              hipStream_t stream) {
  const float* pair   = (const float*)d_in[0];
  const float* single = (const float*)d_in[1];
  const float* Wq = (const float*)d_in[2];  const float* bq = (const float*)d_in[3];
  const float* Wk = (const float*)d_in[4];  const float* bk = (const float*)d_in[5];
  const float* Wv = (const float*)d_in[6];  const float* bv = (const float*)d_in[7];
  const float* Wb = (const float*)d_in[8];  const float* bb = (const float*)d_in[9];
  const float* Wo = (const float*)d_in[10]; const float* bo = (const float*)d_in[11];
  float* out = (float*)d_out;

  uint8_t* ws = (uint8_t*)d_ws;
  half_t* s_h  = (half_t*)ws;        ws += (size_t)SEG_S * 2;
  half_t* wT   = (half_t*)ws;        ws += (size_t)3 * SEG_W * 2;
  half_t* wbT  = (half_t*)ws;        ws += (size_t)SEG_WB * 2;
  half_t* woT  = (half_t*)ws;        ws += (size_t)SEG_WO * 2;
  half_t* q_g  = (half_t*)ws;        ws += (size_t)NH * N_TOK * DH * 2;
  half_t* k_g  = (half_t*)ws;        ws += (size_t)NH * N_TOK * DH * 2;
  half_t* vT_g = (half_t*)ws;        ws += (size_t)NH * N_TOK * DH * 2;
  half_t* o_g  = (half_t*)ws;        ws += (size_t)N_TOK * HD * 2;
  const int nblk = (N_TOK / 16) * JSPLIT;  // 512
  float* o_part = (float*)ws;        ws += (size_t)nblk * NH * 16 * DH * 4;
  float* m_part = (float*)ws;        ws += (size_t)nblk * NH * 16 * 4;
  float* l_part = (float*)ws;        ws += (size_t)nblk * NH * 16 * 4;

  prep_kernel<<<(PREP_TOTAL + 255) / 256, 256, 0, stream>>>(
      single, Wq, Wk, Wv, Wb, Wo, s_h, wT, wbT, woT);

  qkv_kernel<<<dim3(N_TOK / 16, HD / 16, 3), 32, 0, stream>>>(
      s_h, wT, bq, bk, bv, q_g, k_g, vT_g);

  attn_kernel<<<dim3(N_TOK / 16, JSPLIT), 256, 0, stream>>>(
      pair, wbT, bb, q_g, k_g, vT_g, o_part, m_part, l_part);

  combine_kernel<<<(N_TOK * NH * DH + 255) / 256, 256, 0, stream>>>(
      o_part, m_part, l_part, o_g);

  proj_kernel<<<dim3(N_TOK / 16, D_SINGLE / 16), 32, 0, stream>>>(
      o_g, woT, single, bo, out);
}
